// Effi_MVS_57174604645035
// MI455X (gfx1250) — matrix operand order repaired and checked by reading, not compiled, not run on hardware
//
#include <hip/hip_runtime.h>
#include <hip/hip_bf16.h>

#define B_ 2
#define V_ 3
#define C_ 32
#define D_ 32
#define H_ 96
#define W_ 128
#define HW_ (H_ * W_)

typedef __attribute__((ext_vector_type(16))) __bf16 v16bf;
typedef __attribute__((ext_vector_type(8)))  __bf16 v8bf;
typedef __attribute__((ext_vector_type(8)))  float  v8f;

// ---------------------------------------------------------------------------
// 4x4 inverse (adjugate / cofactor expansion)
// ---------------------------------------------------------------------------
__device__ __forceinline__ void inv4(const float m[16], float inv[16]) {
    inv[0]  =  m[5]*m[10]*m[15] - m[5]*m[11]*m[14] - m[9]*m[6]*m[15] + m[9]*m[7]*m[14] + m[13]*m[6]*m[11] - m[13]*m[7]*m[10];
    inv[4]  = -m[4]*m[10]*m[15] + m[4]*m[11]*m[14] + m[8]*m[6]*m[15] - m[8]*m[7]*m[14] - m[12]*m[6]*m[11] + m[12]*m[7]*m[10];
    inv[8]  =  m[4]*m[9]*m[15]  - m[4]*m[11]*m[13] - m[8]*m[5]*m[15] + m[8]*m[7]*m[13] + m[12]*m[5]*m[11] - m[12]*m[7]*m[9];
    inv[12] = -m[4]*m[9]*m[14]  + m[4]*m[10]*m[13] + m[8]*m[5]*m[14] - m[8]*m[6]*m[13] - m[12]*m[5]*m[10] + m[12]*m[6]*m[9];
    inv[1]  = -m[1]*m[10]*m[15] + m[1]*m[11]*m[14] + m[9]*m[2]*m[15] - m[9]*m[3]*m[14] - m[13]*m[2]*m[11] + m[13]*m[3]*m[10];
    inv[5]  =  m[0]*m[10]*m[15] - m[0]*m[11]*m[14] - m[8]*m[2]*m[15] + m[8]*m[3]*m[14] + m[12]*m[2]*m[11] - m[12]*m[3]*m[10];
    inv[9]  = -m[0]*m[9]*m[15]  + m[0]*m[11]*m[13] + m[8]*m[1]*m[15] - m[8]*m[3]*m[13] - m[12]*m[1]*m[11] + m[12]*m[3]*m[9];
    inv[13] =  m[0]*m[9]*m[14]  - m[0]*m[10]*m[13] - m[8]*m[1]*m[14] + m[8]*m[2]*m[13] + m[12]*m[1]*m[10] - m[12]*m[2]*m[9];
    inv[2]  =  m[1]*m[6]*m[15]  - m[1]*m[7]*m[14]  - m[5]*m[2]*m[15] + m[5]*m[3]*m[14] + m[13]*m[2]*m[7]  - m[13]*m[3]*m[6];
    inv[6]  = -m[0]*m[6]*m[15]  + m[0]*m[7]*m[14]  + m[4]*m[2]*m[15] - m[4]*m[3]*m[14] - m[12]*m[2]*m[7]  + m[12]*m[3]*m[6];
    inv[10] =  m[0]*m[5]*m[15]  - m[0]*m[7]*m[13]  - m[4]*m[1]*m[15] + m[4]*m[3]*m[13] + m[12]*m[1]*m[7]  - m[12]*m[3]*m[5];
    inv[14] = -m[0]*m[5]*m[14]  + m[0]*m[6]*m[13]  + m[4]*m[1]*m[14] - m[4]*m[2]*m[13] - m[12]*m[1]*m[6]  + m[12]*m[2]*m[5];
    inv[3]  = -m[1]*m[6]*m[11]  + m[1]*m[7]*m[10]  + m[5]*m[2]*m[11] - m[5]*m[3]*m[10] - m[9]*m[2]*m[7]   + m[9]*m[3]*m[6];
    inv[7]  =  m[0]*m[6]*m[11]  - m[0]*m[7]*m[10]  - m[4]*m[2]*m[11] + m[4]*m[3]*m[10] + m[8]*m[2]*m[7]   - m[8]*m[3]*m[6];
    inv[11] = -m[0]*m[5]*m[11]  + m[0]*m[7]*m[9]   + m[4]*m[1]*m[11] - m[4]*m[3]*m[9]  - m[8]*m[1]*m[7]   + m[8]*m[3]*m[5];
    inv[15] =  m[0]*m[5]*m[10]  - m[0]*m[6]*m[9]   - m[4]*m[1]*m[10] + m[4]*m[2]*m[9]  + m[8]*m[1]*m[6]   - m[8]*m[2]*m[5];
    float det = m[0]*inv[0] + m[1]*inv[4] + m[2]*inv[8] + m[3]*inv[12];
    float id = 1.0f / det;
    #pragma unroll
    for (int i = 0; i < 16; ++i) inv[i] *= id;
}

__device__ __forceinline__ void compose(const float* proj, int b, int v, float out[16]) {
    const float* E = proj + ((size_t)(b * V_ + v) * 2 + 0) * 16;
    const float* K = proj + ((size_t)(b * V_ + v) * 2 + 1) * 16;
    #pragma unroll
    for (int i = 0; i < 16; ++i) out[i] = E[i];
    #pragma unroll
    for (int i = 0; i < 3; ++i)
        #pragma unroll
        for (int j = 0; j < 4; ++j) {
            float s = 0.f;
            #pragma unroll
            for (int k = 0; k < 3; ++k) s += K[i*4+k] * E[k*4+j];
            out[i*4+j] = s;
        }
}

// K0: per (b, src_view) homography rot(9) + trans(3) -> xf[(b*2+v-1)*12]
__global__ void mvs_prep_xform(const float* __restrict__ proj, float* __restrict__ xf) {
    int t = threadIdx.x;
    if (t >= B_ * 2) return;
    int b = t >> 1, v = (t & 1) + 1;
    float compv[16], compr[16], invr[16], T[16];
    compose(proj, b, v, compv);
    compose(proj, b, 0, compr);
    inv4(compr, invr);
    #pragma unroll
    for (int i = 0; i < 4; ++i)
        #pragma unroll
        for (int j = 0; j < 4; ++j) {
            float s = 0.f;
            #pragma unroll
            for (int k = 0; k < 4; ++k) s += compv[i*4+k] * invr[k*4+j];
            T[i*4+j] = s;
        }
    float* o = xf + t * 12;
    o[0]=T[0]; o[1]=T[1]; o[2]=T[2];
    o[3]=T[4]; o[4]=T[5]; o[5]=T[6];
    o[6]=T[8]; o[7]=T[9]; o[8]=T[10];
    o[9]=T[3]; o[10]=T[7]; o[11]=T[11];
}

// K1: plane-sweep warp + variance; store channel-last bf16 volume
// varv[((b*D+d)*HW + h*W + w)*C + c]
__global__ void __launch_bounds__(256)
mvs_build_var(const float* __restrict__ fea, const float* __restrict__ depthv,
              const float* __restrict__ xf, __bf16* __restrict__ varv) {
    int tid = threadIdx.x;
    int w = blockIdx.x * 16 + (tid & 15);
    int h = blockIdx.y * 16 + (tid >> 4);
    int bd = blockIdx.z;
    int b = bd / D_, d = bd % D_;

    float dep = depthv[(size_t)(b * D_ + d) * HW_ + h * W_ + w];

    float wt[2][4];
    int   off[2][4];
    #pragma unroll
    for (int vi = 0; vi < 2; ++vi) {
        const float* X = xf + (b * 2 + vi) * 12;
        float x = (float)w, y = (float)h;
        float rx = X[0]*x + X[1]*y + X[2];
        float ry = X[3]*x + X[4]*y + X[5];
        float rz = X[6]*x + X[7]*y + X[8];
        float pz = rz * dep + X[11];
        float iz = 1.0f / pz;
        float px = (rx * dep + X[9])  * iz;
        float py = (ry * dep + X[10]) * iz;
        float x0 = floorf(px), y0 = floorf(py);
        float fx = px - x0, fy = py - y0;
        #pragma unroll
        for (int tp = 0; tp < 4; ++tp) {
            int dx = tp & 1, dy = tp >> 1;
            float xi = x0 + (float)dx, yi = y0 + (float)dy;
            float wgt = (dx ? fx : 1.f - fx) * (dy ? fy : 1.f - fy);
            bool valid = (xi >= 0.f) && (xi <= (float)(W_ - 1)) && (yi >= 0.f) && (yi <= (float)(H_ - 1));
            wt[vi][tp] = valid ? wgt : 0.f;
            int xc = (int)fminf(fmaxf(xi, 0.f), (float)(W_ - 1));
            int yc = (int)fminf(fmaxf(yi, 0.f), (float)(H_ - 1));
            off[vi][tp] = yc * W_ + xc;
        }
    }

    const float inv3 = 1.0f / 3.0f;
    size_t refbase = (size_t)(b * V_ + 0) * C_ * HW_ + h * W_ + w;
    v8bf pack[4];
    #pragma unroll 4
    for (int c = 0; c < C_; ++c) {
        float r = fea[refbase + (size_t)c * HW_];
        float s = r, q = r * r;
        #pragma unroll
        for (int vi = 0; vi < 2; ++vi) {
            const float* fb = fea + (size_t)((b * V_ + (vi + 1)) * C_ + c) * HW_;
            float smp = wt[vi][0] * fb[off[vi][0]] + wt[vi][1] * fb[off[vi][1]]
                      + wt[vi][2] * fb[off[vi][2]] + wt[vi][3] * fb[off[vi][3]];
            s += smp; q += smp * smp;
        }
        float mu = s * inv3;
        pack[c >> 3][c & 7] = (__bf16)(q * inv3 - mu * mu);
    }
    v8bf* dst = (v8bf*)(varv + ((size_t)(b * D_ + d) * HW_ + h * W_ + w) * C_);
    dst[0] = pack[0]; dst[1] = pack[1]; dst[2] = pack[2]; dst[3] = pack[3];
}

// K2: 3x3x3 conv over C via bf16 WMMA. Per block: one 16x16 output tile at
// fixed (b,d). Outer loop over the 3 dz slices is fully unrolled so the
// B-register set is statically indexed (one live v16bf, no scratch spill);
// inner loop walks the 21 padded M-tiles (18x18 halo rows) of that plane.
// A (16x32 bf16) = var rows; B (32x16) = weights for that dz slice (9 taps).
// D-tile f32 values scatter-accumulate into LDS via ds_add_f32.
__global__ void __launch_bounds__(256)
mvs_conv_wmma(const __bf16* __restrict__ varv, const float* __restrict__ wreg,
              const float* __restrict__ breg, float* __restrict__ pre) {
    __shared__ float lds_pre[256];
    int tid  = threadIdx.x;
    int lane = tid & 31;
    int wave = tid >> 5;
    int w0 = blockIdx.x * 16, h0 = blockIdx.y * 16;
    int bd = blockIdx.z;
    int b = bd / D_, d = bd % D_;

    lds_pre[tid] = 0.f;
    __syncthreads();

    int n    = lane & 15;   // B column / D column = tap index; also A row
    int half = lane >> 4;

    #pragma unroll
    for (int s = 0; s < 3; ++s) {
        // B layout (32x16 K-major, bf16): lane<16 holds K in {0..7,16..23},
        // lane>=16 holds K in {8..15,24..31}. Built per dz slice: statically indexed.
        v16bf bs;
        #pragma unroll
        for (int e = 0; e < 16; ++e) {
            int k = (e & 7) + 8 * half + ((e >> 3) << 4);
            float wv = 0.f;
            if (n < 9) wv = wreg[((k * 3 + s) * 3 + (n / 3)) * 3 + (n % 3)];
            bs[e] = (__bf16)wv;
        }

        int z = d - 1 + s;
        bool zok = (z >= 0) && (z < D_);

        for (int tt = wave; tt < 21; tt += 8) {
            int rowbase = tt * 16;
            int ridx    = rowbase + n;     // halo pixel index 0..323 (18x18)

            v16bf a;
            bool rv = zok && (ridx < 324);
            int hy = ridx / 18, hx = ridx % 18;
            int gh = h0 - 1 + hy, gw = w0 - 1 + hx;
            rv = rv && (gh >= 0) && (gh < H_) && (gw >= 0) && (gw < W_);
            if (rv) {
                // A 16-bit layout: lane<16 -> K in {0..7,16..23}; lane>=16 -> {8..15,24..31}
                const __bf16* base = varv + ((size_t)(b * D_ + z) * HW_ + gh * W_ + gw) * C_;
                // hint the rows this wave touches ~8 tiles (≈7 image rows) later
                __builtin_prefetch(base + 7 * W_ * C_, 0, 1);
                v8bf lo = *(const v8bf*)(base + half * 8);
                v8bf hi = *(const v8bf*)(base + 16 + half * 8);
                #pragma unroll
                for (int e = 0; e < 8; ++e) { a[e] = lo[e]; a[8 + e] = hi[e]; }
            } else {
                #pragma unroll
                for (int e = 0; e < 16; ++e) a[e] = (__bf16)0.f;
            }

            v8f acc = {};
            acc = __builtin_amdgcn_wmma_f32_16x16x32_bf16(
                /*neg_a=*/false, a, /*neg_b=*/false, bs,
                /*c_mod=*/(short)0, acc, /*reuse_a=*/false, /*reuse_b=*/false);

            if (n < 9) {
                int dy = n / 3, dx = n % 3;   // tap offsets (0..2); out = halo - offset
                #pragma unroll
                for (int r = 0; r < 8; ++r) {
                    int mr = r + (half ? 8 : 0);        // D row held in this VGPR
                    int rg = rowbase + mr;
                    if (rg < 324) {
                        int hyr = rg / 18, hxr = rg % 18;
                        int oy = hyr - dy, ox = hxr - dx;
                        if (oy >= 0 && oy < 16 && ox >= 0 && ox < 16)
                            atomicAdd(&lds_pre[oy * 16 + ox], acc[r]);
                    }
                }
            }
        }
    }
    __syncthreads();

    int oy = tid >> 4, ox = tid & 15;
    pre[(size_t)(b * D_ + d) * HW_ + (h0 + oy) * W_ + (w0 + ox)] = lds_pre[tid] + breg[0];
}

// K3: softmax over D, expected depth, 4-window confidence
__global__ void __launch_bounds__(256)
mvs_finalize(const float* __restrict__ pre, const float* __restrict__ depthv,
             float* __restrict__ outp) {
    int g = blockIdx.x * 256 + threadIdx.x;
    if (g >= B_ * HW_) return;
    int b = g / HW_, pix = g % HW_;

    float p[D_];
    float mx = -3.4e38f;
    #pragma unroll
    for (int dd = 0; dd < D_; ++dd) {
        p[dd] = pre[(size_t)(b * D_ + dd) * HW_ + pix];
        mx = fmaxf(mx, p[dd]);
    }
    float sum = 0.f, sd = 0.f, si = 0.f;
    #pragma unroll
    for (int dd = 0; dd < D_; ++dd) {
        float e = __expf(p[dd] - mx);
        p[dd] = e;
        sum += e;
        sd  += e * depthv[(size_t)(b * D_ + dd) * HW_ + pix];
        si  += e * (float)dd;
    }
    float isum = 1.0f / sum;
    float depth = sd * isum;
    int idx = (int)(si * isum);                 // truncation, matches astype(int32)
    idx = idx < 0 ? 0 : (idx > D_ - 1 ? D_ - 1 : idx);
    float conf = 0.f;
    #pragma unroll
    for (int k = -1; k <= 2; ++k) {
        int kk = idx + k;
        if (kk >= 0 && kk < D_) conf += p[kk];
    }
    conf *= isum;
    outp[g] = depth;
    outp[B_ * HW_ + g] = conf;
}

extern "C" void kernel_launch(void* const* d_in, const int* in_sizes, int n_in,
                              void* d_out, int out_size, void* d_ws, size_t ws_size,
                              hipStream_t stream) {
    (void)in_sizes; (void)n_in; (void)out_size; (void)ws_size;
    const float* fea    = (const float*)d_in[0];   // (B,V,C,H,W)
    const float* proj   = (const float*)d_in[1];   // (B,V,2,4,4)
    const float* depthv = (const float*)d_in[2];   // (B,D,H,W)
    const float* wreg   = (const float*)d_in[3];   // (1,C,3,3,3)
    const float* breg   = (const float*)d_in[4];   // (1,)
    float* out = (float*)d_out;                    // depth (B,H,W) ++ conf (B,H,W)

    char* ws = (char*)d_ws;
    float*  xf   = (float*)ws;                                   // 48 floats
    __bf16* varv = (__bf16*)(ws + 256);                          // 48 MiB bf16 volume
    size_t  var_bytes = (size_t)B_ * D_ * HW_ * C_ * 2;
    float*  pre  = (float*)(ws + 256 + var_bytes);               // 3 MiB

    mvs_prep_xform<<<1, 32, 0, stream>>>(proj, xf);
    dim3 grid(W_ / 16, H_ / 16, B_ * D_);
    mvs_build_var<<<grid, 256, 0, stream>>>(fea, depthv, xf, varv);
    mvs_conv_wmma<<<grid, 256, 0, stream>>>(varv, wreg, breg, pre);
    int np = B_ * HW_;
    mvs_finalize<<<(np + 255) / 256, 256, 0, stream>>>(pre, depthv, out);
}
